// GPT2Block_11510512353461
// MI455X (gfx1250) — compile-verified, matrix operand order checked by reading, not run on hardware
//
#include <hip/hip_runtime.h>
#include <hip/hip_bf16.h>
#include <math.h>

// ---------- types ----------
typedef __bf16 bf16_t;
typedef __bf16 v8bf  __attribute__((ext_vector_type(8)));
typedef __bf16 v16bf __attribute__((ext_vector_type(16)));
typedef float  v8f   __attribute__((ext_vector_type(8)));
typedef float  v4f   __attribute__((ext_vector_type(4)));

#define WMMA_BF16(A, B, C) \
  __builtin_amdgcn_wmma_f32_16x16x32_bf16(false, (A), false, (B), (short)0, (C), false, false)

__device__ __forceinline__ v16bf cat8(v8bf lo, v8bf hi) {
  return __builtin_shufflevector(lo, hi, 0,1,2,3,4,5,6,7,8,9,10,11,12,13,14,15);
}
__device__ __forceinline__ v8f zero_v8f() {
  v8f z;
#pragma unroll
  for (int i = 0; i < 8; ++i) z[i] = 0.0f;
  return z;
}
__device__ __forceinline__ v8bf zero_v8bf() {
  v8bf z;
#pragma unroll
  for (int i = 0; i < 8; ++i) z[i] = (bf16_t)0.0f;
  return z;
}

// ---------- problem constants ----------
#define BATCH 8
#define SEQ   1024
#define EMB   1024
#define HEADS 16
#define TKV   196
#define TKV_P 224   // padded cross kv length (multiple of 32)
#define HD    64

// =====================================================================
// GEMM: C[M,N] = A[M,K] (bf16) @ W[N,K]^T (fp32, converted) + bias
// mode 0: out bf16 (optional exact GELU) ; mode 1: out fp32 = acc + bias + res
// Block tile: 128(M) x 64(N), K-step 64, 256 threads (8 waves),
// double-buffered LDS (one barrier per K-step, global loads overlapped
// with WMMA), wave w computes rows [w*16, w*16+16) x all 64 cols.
// =====================================================================
#define GBM 128
#define GBN 64
#define GBK 64
#define LDP 72   // padded LDS row stride (bf16 elems): 36 banks, conflict-free

__global__ __launch_bounds__(256) void gemm_wmma(
    const bf16_t* __restrict__ A, int lda,
    const float*  __restrict__ W, int ldw,
    const float*  __restrict__ bias,
    bf16_t* __restrict__ outb,
    float*  __restrict__ outf,
    const float* __restrict__ res,
    int M, int N, int K, int ldo, int mode, int gelu) {
  __shared__ bf16_t sA[2][GBM * LDP];
  __shared__ bf16_t sW[2][GBN * LDP];

  const int tid  = threadIdx.x;
  const int wave = tid >> 5;
  const int lane = tid & 31;
  const int lh   = lane & 15;
  const int lu   = lane >> 4;

  const int m0 = blockIdx.y * GBM;
  const int n0 = blockIdx.x * GBN;

  v8f acc[4];
#pragma unroll
  for (int j = 0; j < 4; ++j) acc[j] = zero_v8f();

  // staging maps: A 128x64 bf16 (32 elems/thread), W 64x64 fp32->bf16 (16/thread)
  const int ar = tid >> 1;
  const int ac = (tid & 1) * 32;
  const int wr = tid >> 2;
  const int wc = (tid & 3) * 16;

  const int gr = m0 + ar;
  const bf16_t* Arow = A + (size_t)gr * lda + ac;
  const float*  Wrow = W + (size_t)(n0 + wr) * ldw + wc;

  v8bf areg[4], wreg[2];

  auto load_tile = [&](int k0) {
    if (gr < M) {
      const bf16_t* p = Arow + k0;
#pragma unroll
      for (int i = 0; i < 4; ++i) areg[i] = *(const v8bf*)(p + i * 8);
    } else {
#pragma unroll
      for (int i = 0; i < 4; ++i) areg[i] = zero_v8bf();
    }
    const float* q = Wrow + k0;
#pragma unroll
    for (int i = 0; i < 2; ++i) {
      v4f w0 = *(const v4f*)(q + i * 8);
      v4f w1 = *(const v4f*)(q + i * 8 + 4);
      v8bf wb;
#pragma unroll
      for (int j = 0; j < 4; ++j) { wb[j] = (bf16_t)w0[j]; wb[4 + j] = (bf16_t)w1[j]; }
      wreg[i] = wb;
    }
  };
  auto store_tile = [&](int buf) {
#pragma unroll
    for (int i = 0; i < 4; ++i)
      *(v8bf*)(&sA[buf][ar * LDP + ac + i * 8]) = areg[i];
#pragma unroll
    for (int i = 0; i < 2; ++i)
      *(v8bf*)(&sW[buf][wr * LDP + wc + i * 8]) = wreg[i];
  };

  load_tile(0);
  store_tile(0);

  const int nk = K / GBK;
  for (int kt = 0; kt < nk; ++kt) {
    __syncthreads();                       // buf[cur] visible to all waves
    const int cur = kt & 1;
    if (kt + 1 < nk) load_tile((kt + 1) * GBK);   // overlap with WMMA below
    if (kt + 2 < nk) {                            // L2->near prefetch, tile after next
      __builtin_prefetch(Wrow + (kt + 2) * GBK, 0, 0);
      if (gr < M) __builtin_prefetch(Arow + (kt + 2) * GBK, 0, 0);
    }

    const bf16_t* arow = &sA[cur][(wave * 16 + lh) * LDP];
#pragma unroll
    for (int kh = 0; kh < 2; ++kh) {
      v16bf af = cat8(*(const v8bf*)(arow + kh * 32 + lu * 8),
                      *(const v8bf*)(arow + kh * 32 + 16 + lu * 8));
      v16bf bf4[4];
#pragma unroll
      for (int j = 0; j < 4; ++j) {
        const bf16_t* wrow = &sW[cur][(j * 16 + lh) * LDP];
        bf4[j] = cat8(*(const v8bf*)(wrow + kh * 32 + lu * 8),
                      *(const v8bf*)(wrow + kh * 32 + 16 + lu * 8));
      }
#pragma unroll
      for (int j = 0; j < 4; ++j) acc[j] = WMMA_BF16(af, bf4[j], acc[j]);
    }
    if (kt + 1 < nk) store_tile(1 - cur);  // disjoint from buf[cur] reads
  }

  // ---- epilogue ----
#pragma unroll
  for (int j = 0; j < 4; ++j) {
    const int n = n0 + j * 16 + lh;
    const float bv = bias[n];
#pragma unroll
    for (int r = 0; r < 8; ++r) {
      const int m = m0 + wave * 16 + r + 8 * lu;
      if (m < M) {
        float v = acc[j][r] + bv;
        if (gelu) v = 0.5f * v * (1.0f + erff(v * 0.70710678118f));
        const size_t o = (size_t)m * ldo + n;
        if (mode == 0) outb[o] = (bf16_t)v;
        else           outf[o] = v + res[o];
      }
    }
  }
}

// =====================================================================
// Flash attention (transposed WMMA): scores^T = K(A) x Q^T(B),
// O^T = V^T(A) x P^T(B). Per-wave 16 queries; softmax state per lane,
// merged across the (lane, lane^16) pair which shares a query.
// =====================================================================
__global__ __launch_bounds__(128) void attn_wmma(
    const bf16_t* __restrict__ qp, int qstride, long long q_bstr,
    const bf16_t* __restrict__ kp, int kstride, long long k_bstr,
    const bf16_t* __restrict__ vtp, int vtstride,
    bf16_t* __restrict__ op, int ostride, long long o_bstr,
    int kvlen, int causal) {
  const int bh   = blockIdx.y;
  const int b    = bh / HEADS;
  const int h    = bh % HEADS;
  const int wave = threadIdx.x >> 5;
  const int lane = threadIdx.x & 31;
  const int lh   = lane & 15;
  const int lu   = lane >> 4;
  const int q0   = blockIdx.x * 64 + wave * 16;

  const bf16_t* qb = qp + (size_t)b * q_bstr + h * HD;
  const bf16_t* kb = kp + (size_t)b * k_bstr + h * HD;
  const bf16_t* vb = vtp + (size_t)bh * HD * vtstride;
  bf16_t*       ob = op + (size_t)b * o_bstr + h * HD;

  // Q^T B-fragments (head dim 64 -> 2 frags); lane holds query column q0+lh
  const bf16_t* qrow = qb + (size_t)(q0 + lh) * qstride;
  v16bf qf[2];
#pragma unroll
  for (int f = 0; f < 2; ++f)
    qf[f] = cat8(*(const v8bf*)(qrow + f * 32 + lu * 8),
                 *(const v8bf*)(qrow + f * 32 + 16 + lu * 8));

  v8f o_[4];
#pragma unroll
  for (int f = 0; f < 4; ++f) o_[f] = zero_v8f();
  float m_run = -INFINITY, l_run = 0.0f;
  const float scale = 0.125f;  // 1/sqrt(64)
  const int qidx = q0 + lh;

  const int kmax   = causal ? (q0 + 16) : kvlen;
  const int ntiles = (kmax + 31) >> 5;

  for (int kt = 0; kt < ntiles; ++kt) {
    const int k0 = kt * 32;
    // ---- scores^T: two 16-key C-frags, K-dim = head dim (2 wmma each) ----
    v8f st[2];
    st[0] = zero_v8f(); st[1] = zero_v8f();
#pragma unroll
    for (int s = 0; s < 2; ++s) {
      const bf16_t* krow = kb + (size_t)(k0 + s * 16 + lh) * kstride;
#pragma unroll
      for (int dh = 0; dh < 2; ++dh) {
        v16bf kf = cat8(*(const v8bf*)(krow + dh * 32 + lu * 8),
                        *(const v8bf*)(krow + dh * 32 + 16 + lu * 8));
        st[s] = WMMA_BF16(kf, qf[dh], st[s]);
      }
    }
    // ---- mask + online softmax (per-lane; merge with partner lane^16) ----
    float sc[16];
    float m_new = m_run;
#pragma unroll
    for (int s = 0; s < 2; ++s)
#pragma unroll
      for (int r = 0; r < 8; ++r) {
        const int key = k0 + s * 16 + r + 8 * lu;
        float v = st[s][r] * scale;
        if (key >= kvlen || (causal && key > qidx)) v = -INFINITY;
        sc[s * 8 + r] = v;
        m_new = fmaxf(m_new, v);
      }
    m_new = fmaxf(m_new, __shfl_xor(m_new, 16));
    const float corr = __expf(m_run - m_new);
    float psum = 0.0f;
    v16bf pf;  // P^T B-frag: e<8 -> keys of st[0], e>=8 -> keys of st[1]
#pragma unroll
    for (int i = 0; i < 16; ++i) {
      const float p = __expf(sc[i] - m_new);
      psum += p;
      pf[i] = (bf16_t)p;
    }
    l_run = l_run * corr + psum;
    m_run = m_new;
#pragma unroll
    for (int f = 0; f < 4; ++f) o_[f] = o_[f] * corr;
    // ---- O^T += V^T x P^T (4 d-frags) ----
#pragma unroll
    for (int f = 0; f < 4; ++f) {
      const bf16_t* vrow = vb + (size_t)(f * 16 + lh) * vtstride + k0;
      v16bf vf = cat8(*(const v8bf*)(vrow + lu * 8),
                      *(const v8bf*)(vrow + 16 + lu * 8));
      o_[f] = WMMA_BF16(vf, pf, o_[f]);
    }
  }

  const float l_tot = l_run + __shfl_xor(l_run, 16);
  const float inv = 1.0f / l_tot;
  bf16_t* orow = ob + (size_t)(q0 + lh) * ostride;
#pragma unroll
  for (int f = 0; f < 4; ++f)
#pragma unroll
    for (int r = 0; r < 8; ++r)
      orow[f * 16 + r + 8 * lu] = (bf16_t)(o_[f][r] * inv);
}

// =====================================================================
// LayerNorm (fp32 in, bf16 out), one row per block
// =====================================================================
__global__ __launch_bounds__(256) void ln_bf16(
    const float* __restrict__ x, const float* __restrict__ g,
    const float* __restrict__ bb, bf16_t* __restrict__ out) {
  const int row = blockIdx.x;
  const float* xr = x + (size_t)row * EMB;
  float s = 0.0f, s2 = 0.0f;
  for (int i = threadIdx.x; i < EMB; i += 256) {
    const float v = xr[i];
    s += v; s2 += v * v;
  }
#pragma unroll
  for (int off = 16; off > 0; off >>= 1) {
    s  += __shfl_down(s, off);
    s2 += __shfl_down(s2, off);
  }
  __shared__ float red[18];
  const int wave = threadIdx.x >> 5, lane = threadIdx.x & 31;
  if (lane == 0) { red[wave] = s; red[8 + wave] = s2; }
  __syncthreads();
  if (threadIdx.x == 0) {
    float a = 0.0f, b2 = 0.0f;
    for (int w = 0; w < 8; ++w) { a += red[w]; b2 += red[8 + w]; }
    red[16] = a; red[17] = b2;
  }
  __syncthreads();
  const float mu  = red[16] / EMB;
  const float var = red[17] / EMB - mu * mu;
  const float rs  = rsqrtf(var + 1e-5f);
  bf16_t* orow = out + (size_t)row * EMB;
  for (int i = threadIdx.x; i < EMB; i += 256)
    orow[i] = (bf16_t)((xr[i] - mu) * rs * g[i] + bb[i]);
}

// ---------- small elementwise kernels ----------
__global__ void copy_f32_k(const float* __restrict__ a, float* __restrict__ o, long long n) {
  const long long i = (long long)blockIdx.x * 256 + threadIdx.x;
  if (i < n) o[i] = a[i];
}
__global__ void f32_to_bf16_k(const float* __restrict__ a, bf16_t* __restrict__ o, long long n) {
  const long long i = (long long)blockIdx.x * 256 + threadIdx.x;
  if (i < n) o[i] = (bf16_t)a[i];
}
// V transpose (self): vt[(b*H+h)*64 + d][s] = qkv[b*S+s][2048 + h*64 + d]
__global__ void vt_self_k(const bf16_t* __restrict__ qkv, bf16_t* __restrict__ vt) {
  long long i = (long long)blockIdx.x * 256 + threadIdx.x;
  if (i >= (long long)BATCH * HEADS * HD * SEQ) return;
  const int d = (int)(i & 63); i >>= 6;
  const int s = (int)(i & (SEQ - 1)); i >>= 10;
  const int h = (int)(i & (HEADS - 1));
  const int b = (int)(i >> 4);
  vt[((size_t)(b * HEADS + h) * HD + d) * SEQ + s] =
      qkv[(size_t)(b * SEQ + s) * (3 * EMB) + 2 * EMB + h * HD + d];
}
// V transpose (cross) with zero-padded tail: vt stride TKV_P
__global__ void vt_cross_k(const bf16_t* __restrict__ v, bf16_t* __restrict__ vt) {
  long long i = (long long)blockIdx.x * 256 + threadIdx.x;
  if (i >= (long long)BATCH * HEADS * HD * TKV_P) return;
  const int d = (int)(i & 63); i >>= 6;
  const int t = (int)(i % TKV_P); i /= TKV_P;
  const int h = (int)(i & (HEADS - 1));
  const int b = (int)(i >> 4);
  const bf16_t val = (t < TKV)
      ? v[(size_t)(b * TKV + t) * EMB + h * HD + d] : (bf16_t)0.0f;
  vt[((size_t)(b * HEADS + h) * HD + d) * TKV_P + t] = val;
}

// =====================================================================
// launcher
// =====================================================================
extern "C" void kernel_launch(void* const* d_in, const int* in_sizes, int n_in,
                              void* d_out, int out_size, void* d_ws, size_t ws_size,
                              hipStream_t stream) {
  (void)in_sizes; (void)n_in; (void)out_size; (void)ws_size;
  const float* x      = (const float*)d_in[0];
  const float* enc    = (const float*)d_in[1];
  const float* ln1g   = (const float*)d_in[2];
  const float* ln1b   = (const float*)d_in[3];
  const float* qkv_w  = (const float*)d_in[4];
  const float* qkv_b  = (const float*)d_in[5];
  const float* aproj_w= (const float*)d_in[6];
  const float* aproj_b= (const float*)d_in[7];
  const float* ln2g   = (const float*)d_in[8];
  const float* ln2b   = (const float*)d_in[9];
  const float* q_w    = (const float*)d_in[10];
  const float* q_b    = (const float*)d_in[11];
  const float* k_w    = (const float*)d_in[12];
  const float* k_b    = (const float*)d_in[13];
  const float* v_w    = (const float*)d_in[14];
  const float* v_b    = (const float*)d_in[15];
  const float* cproj_w= (const float*)d_in[16];
  const float* cproj_b= (const float*)d_in[17];
  const float* ln3g   = (const float*)d_in[18];
  const float* ln3b   = (const float*)d_in[19];
  const float* fc_w   = (const float*)d_in[20];
  const float* fc_b   = (const float*)d_in[21];
  const float* proj_w = (const float*)d_in[22];
  const float* proj_b = (const float*)d_in[23];
  float* out = (float*)d_out;

  const int ROWS = BATCH * SEQ;        // 8192
  const int EROWS = BATCH * TKV;       // 1568

  // ---- workspace carve-up (~137 MB total) ----
  char* ws = (char*)d_ws;
  size_t off = 0;
  auto carve = [&](size_t bytes) {
    void* p = ws + off;
    off = (off + bytes + 255) & ~(size_t)255;
    return p;
  };
  float*  x_res = (float*) carve((size_t)ROWS * EMB * 4);          // 33.5 MB
  bf16_t* hbuf  = (bf16_t*)carve((size_t)ROWS * EMB * 2);          // 16.8 MB
  char*   regC  = (char*)  carve((size_t)ROWS * 4 * EMB * 2);      // 67.1 MB (reused)
  bf16_t* vtbuf = (bf16_t*)carve((size_t)ROWS * EMB * 2);          // 16.8 MB
  bf16_t* encbf = (bf16_t*)carve((size_t)EROWS * EMB * 2);         // 3.2 MB

  const dim3 blk256(256), blk128(128);
  auto gemm_grid = [](int M, int N) { return dim3(N / GBN, (M + GBM - 1) / GBM); };

  // 0) residual init
  copy_f32_k<<<(ROWS * EMB) / 256, blk256, 0, stream>>>(x, x_res, (long long)ROWS * EMB);

  // ---------------- causal self-attention ----------------
  ln_bf16<<<ROWS, blk256, 0, stream>>>(x_res, ln1g, ln1b, hbuf);

  bf16_t* qkv  = (bf16_t*)regC;                                    // [8192, 3072]
  bf16_t* aout = (bf16_t*)(regC + (size_t)ROWS * 3 * EMB * 2);     // [8192, 1024]
  gemm_wmma<<<gemm_grid(ROWS, 3 * EMB), blk256, 0, stream>>>(
      hbuf, EMB, qkv_w, EMB, qkv_b, qkv, nullptr, nullptr,
      ROWS, 3 * EMB, EMB, 3 * EMB, 0, 0);

  vt_self_k<<<(BATCH * HEADS * HD * SEQ) / 256, blk256, 0, stream>>>(qkv, vtbuf);

  attn_wmma<<<dim3(SEQ / 64, BATCH * HEADS), blk128, 0, stream>>>(
      qkv, 3 * EMB, (long long)SEQ * 3 * EMB,
      qkv + EMB, 3 * EMB, (long long)SEQ * 3 * EMB,
      vtbuf, SEQ,
      aout, EMB, (long long)SEQ * EMB,
      SEQ, 1);

  gemm_wmma<<<gemm_grid(ROWS, EMB), blk256, 0, stream>>>(
      aout, EMB, aproj_w, EMB, aproj_b, nullptr, x_res, x_res,
      ROWS, EMB, EMB, EMB, 1, 0);

  // ---------------- cross-attention ----------------
  ln_bf16<<<ROWS, blk256, 0, stream>>>(x_res, ln2g, ln2b, hbuf);

  bf16_t* qbuf = (bf16_t*)regC;                                          // [8192,1024]
  bf16_t* kbuf = (bf16_t*)(regC + (size_t)ROWS * EMB * 2);               // [1568,1024]
  bf16_t* vbuf = (bf16_t*)(regC + (size_t)(ROWS + EROWS) * EMB * 2);     // [1568,1024]
  bf16_t* aout2= (bf16_t*)(regC + (size_t)(ROWS + 2 * EROWS) * EMB * 2); // [8192,1024]

  gemm_wmma<<<gemm_grid(ROWS, EMB), blk256, 0, stream>>>(
      hbuf, EMB, q_w, EMB, q_b, qbuf, nullptr, nullptr,
      ROWS, EMB, EMB, EMB, 0, 0);

  f32_to_bf16_k<<<(EROWS * EMB + 255) / 256, blk256, 0, stream>>>(
      enc, encbf, (long long)EROWS * EMB);

  gemm_wmma<<<gemm_grid(EROWS, EMB), blk256, 0, stream>>>(
      encbf, EMB, k_w, EMB, k_b, kbuf, nullptr, nullptr,
      EROWS, EMB, EMB, EMB, 0, 0);
  gemm_wmma<<<gemm_grid(EROWS, EMB), blk256, 0, stream>>>(
      encbf, EMB, v_w, EMB, v_b, vbuf, nullptr, nullptr,
      EROWS, EMB, EMB, EMB, 0, 0);

  vt_cross_k<<<(BATCH * HEADS * HD * TKV_P) / 256, blk256, 0, stream>>>(vbuf, vtbuf);

  attn_wmma<<<dim3(SEQ / 64, BATCH * HEADS), blk128, 0, stream>>>(
      qbuf, EMB, (long long)SEQ * EMB,
      kbuf, EMB, (long long)TKV * EMB,
      vtbuf, TKV_P,
      aout2, EMB, (long long)SEQ * EMB,
      TKV, 0);

  gemm_wmma<<<gemm_grid(ROWS, EMB), blk256, 0, stream>>>(
      aout2, EMB, cproj_w, EMB, cproj_b, nullptr, x_res, x_res,
      ROWS, EMB, EMB, EMB, 1, 0);

  // ---------------- MLP ----------------
  ln_bf16<<<ROWS, blk256, 0, stream>>>(x_res, ln3g, ln3b, hbuf);

  bf16_t* fcbuf = (bf16_t*)regC;                                   // [8192, 4096]
  gemm_wmma<<<gemm_grid(ROWS, 4 * EMB), blk256, 0, stream>>>(
      hbuf, EMB, fc_w, EMB, fc_b, fcbuf, nullptr, nullptr,
      ROWS, 4 * EMB, EMB, 4 * EMB, 0, 1);

  gemm_wmma<<<gemm_grid(ROWS, EMB), blk256, 0, stream>>>(
      fcbuf, 4 * EMB, proj_w, 4 * EMB, proj_b, nullptr, out, x_res,
      ROWS, EMB, 4 * EMB, EMB, 1, 0);
}
